// RobustGSA_31817117729249
// MI455X (gfx1250) — matrix operand order repaired and checked by reading, not compiled, not run on hardware
//
#include <hip/hip_runtime.h>
#include <stdint.h>

// Problem constants (from reference)
#define Bc 4
#define Tc 1024
#define Dc 512
#define Hc 8
#define Sc 16
#define HDc 64

typedef __attribute__((ext_vector_type(16))) __bf16 v16bf;
typedef __attribute__((ext_vector_type(8)))  float  v8f;

union AB16 {
    v16bf    v;
    __bf16   e[16];
    uint32_t u[8];
};

// ---------------------------------------------------------------------------
// Generic C(MxN) = A(MxK) @ W(NxK)^T, fp32 in/out, bf16 WMMA compute.
// One wave per 16x16 C tile; K stepped by 32 (one v_wmma_f32_16x16x32_bf16
// per step). A/W fragments gathered in contiguous k order (same for A and B):
//   A (16x32 bf16): lane m=lane&15 holds row m; u[0..7]=K(16*half+0..15) pairs.
//   B (32x16 bf16): lane n=lane&15 holds col n; u[0..7]=K(16*half+0..15) pairs.
// W is row-major (N x K), so B column n == W row (n0+n): contiguous loads.
// ---------------------------------------------------------------------------
__global__ __launch_bounds__(256) void gemm_bf16_kernel(
    const float* __restrict__ A, const float* __restrict__ W,
    float* __restrict__ C, int M, int N, int K)
{
    const int wid  = blockIdx.x * 8 + (threadIdx.x >> 5);
    const int lane = threadIdx.x & 31;
    const int tiles_n = N >> 4;
    if (wid >= (M >> 4) * tiles_n) return;
    const int i0 = (wid / tiles_n) << 4;
    const int n0 = (wid % tiles_n) << 4;
    const int n    = lane & 15;
    const int half = lane >> 4;

    const float* arow = A + (size_t)(i0 + n) * K;  // A row for this lane (m = lane&15)
    const float* wrow = W + (size_t)(n0 + n) * K;  // W row = B column for this lane

    v8f acc = {0.f, 0.f, 0.f, 0.f, 0.f, 0.f, 0.f, 0.f};

    for (int k0 = 0; k0 < K; k0 += 32) {
        AB16 af, bf;
        const float* ap = arow + k0 + 16 * half;
        const float* wp = wrow + k0 + 16 * half;
        __builtin_prefetch(ap + 32, 0, 0);  // global_prefetch_b8 next K chunk
#pragma unroll
        for (int t = 0; t < 8; ++t) af.e[t]     = (__bf16)ap[t];
#pragma unroll
        for (int t = 0; t < 8; ++t) af.e[8 + t] = (__bf16)ap[8 + t];
#pragma unroll
        for (int t = 0; t < 16; ++t) bf.e[t]    = (__bf16)wp[t];
        acc = __builtin_amdgcn_wmma_f32_16x16x32_bf16(
            false, af.v, false, bf.v, (short)0, acc, false, false);
    }
#pragma unroll
    for (int v = 0; v < 8; ++v)
        C[(size_t)(i0 + v + 8 * half) * N + (n0 + n)] = acc[v];
}

// ---------------------------------------------------------------------------
// Gaussian features: q_w,k_w = exp(-0.5*min(d2,100)*inv_var)*eff_amp, stored
// bf16 [B,H,T,S] (logits fragments are then contiguous loads). Also transposes
// V to bf16 [B,H,HD,T] so AV B-fragments are contiguous 32B loads.
// One block per token (128 threads = one (h,s) each).
// ---------------------------------------------------------------------------
__global__ __launch_bounds__(128) void features_kernel(
    const float* __restrict__ qkv, const float* __restrict__ base_centers,
    const float* __restrict__ center_deltas, const float* __restrict__ log_scales,
    const float* __restrict__ log_amps, const float* __restrict__ movement_param,
    __bf16* __restrict__ qwb, __bf16* __restrict__ kwb, __bf16* __restrict__ vtb)
{
    const int bt = blockIdx.x;                // b*T + t
    const int b  = bt >> 10;                  // /Tc
    const int t  = bt & (Tc - 1);
    const int tid = threadIdx.x;              // 0..127
    const int h = tid >> 4, s = tid & 15;

    const float move = 0.2f / (1.f + __expf(-movement_param[0]));
    float sc  = __expf(log_scales[h * Sc + s]);
    sc = fminf(fmaxf(sc, 1e-4f), 5.f);
    float amp = __expf(log_amps[h * Sc + s]);
    amp = fminf(fmaxf(amp, 1e-6f), 10.f);
    const float eff = (amp > 0.001f) ? amp : 0.f;
    const float inv_var = 1.f / (sc * sc + 1e-8f);

    const float* qp = qkv + (size_t)bt * (3 * Dc) + h * HDc;           // q slice
    const float* kp = qp + Dc;                                         // k slice
    const float* cp = base_centers  + (size_t)(h * Sc + s) * HDc;
    const float* dp = center_deltas + (size_t)(h * Sc + s) * HDc;

    float qd = 0.f, kd = 0.f;
#pragma unroll 8
    for (int d = 0; d < HDc; ++d) {
        const float c  = cp[d] + dp[d] * move;
        const float dq = qp[d] - c;
        const float dk = kp[d] - c;
        qd += dq * dq;
        kd += dk * dk;
    }
    qd = fminf(qd, 100.f);
    kd = fminf(kd, 100.f);
    const float qw = __expf(-0.5f * qd * inv_var) * eff;
    const float kw = __expf(-0.5f * kd * inv_var) * eff;
    const size_t widx = (((size_t)b * Hc + h) * Tc + t) * Sc + s;
    qwb[widx] = (__bf16)qw;
    kwb[widx] = (__bf16)kw;

    // V transpose: 512 values per token, 4 per thread -> vt[b][h][hd][t]
#pragma unroll
    for (int r = 0; r < 4; ++r) {
        const int c  = tid * 4 + r;
        const int hh = c >> 6, hd = c & 63;
        const float v = qkv[(size_t)bt * (3 * Dc) + 2 * Dc + c];
        vtb[(((size_t)b * Hc + hh) * HDc + hd) * Tc + t] = (__bf16)v;
    }
}

// Rank-16 logits tile: one bf16 WMMA (K padded 16->32 with zeros in registers).
__device__ __forceinline__ v8f logits_tile(const AB16& af, const __bf16* kbase,
                                           int j0, int n, int half)
{
    AB16 bf;
    if (half == 0) {
        const uint32_t* kr = (const uint32_t*)(kbase + (size_t)(j0 + n) * Sc);
#pragma unroll
        for (int v = 0; v < 8; ++v) bf.u[v] = kr[v];   // K=0..15 = full S row
    } else {
#pragma unroll
        for (int v = 0; v < 8; ++v) bf.u[v] = 0u;      // K=16..31 zero pad
    }
    const v8f z = {0.f, 0.f, 0.f, 0.f, 0.f, 0.f, 0.f, 0.f};
    return __builtin_amdgcn_wmma_f32_16x16x32_bf16(
        false, af.v, false, bf.v, (short)0, z, false, false);
}

// ---------------------------------------------------------------------------
// Fused attention: per-(b, i-tile) block, 8 waves = 8 heads. 3 passes over j
// with rank-16 logits recompute (1 WMMA per 16x16 tile):
//   pass1: row max; pass2: row sum of exp; pass3: attn write + attn@V.
// Pass-3 attn tile (16 rows x 32 j x 8 h fp32, matching the global (i,j,h)
// layout exactly) is staged in LDS and drained with coalesced 128-bit
// *async* LDS->global stores (ASYNCcnt path), keeping the dominant 134MB
// stream off the VALU. attn->A-fragment relayout for AV uses a 1KB/wave
// bf16 LDS tile.
// ---------------------------------------------------------------------------
__global__ __launch_bounds__(256) void attn_kernel(
    const __bf16* __restrict__ qwb, const __bf16* __restrict__ kwb,
    const __bf16* __restrict__ vtb, const float* __restrict__ temperature,
    float* __restrict__ attn, float* __restrict__ Oc)
{
    __shared__ __bf16 pshared[8][16 * 32];       // 8 KB: per-wave P tiles (bf16)
    __shared__ float  attn_stage[16 * 32 * 8];   // 16 KB: [row][j][h] fp32

    const int wave = threadIdx.x >> 5;
    const int lane = threadIdx.x & 31;
    const int n    = lane & 15;
    const int half = lane >> 4;
    const int b  = blockIdx.x / (Tc / 16);
    const int i0 = (blockIdx.x % (Tc / 16)) << 4;
    const int h  = wave;

    float temp = temperature[0];
    temp = fminf(fmaxf(temp, 0.01f), 10.f);
    const float inv_temp = 1.f / temp;

    const __bf16* qbase = qwb + ((size_t)b * Hc + h) * Tc * Sc;
    const __bf16* kbase = kwb + ((size_t)b * Hc + h) * Tc * Sc;
    const __bf16* vbase = vtb + ((size_t)b * Hc + h) * HDc * Tc;

    // q_w A-fragment (rows i0..i0+15, K = S padded to 32)
    AB16 af;
    {
        const uint32_t* qr =
            (const uint32_t*)(qbase + (size_t)(i0 + n) * Sc);
#pragma unroll
        for (int v = 0; v < 8; ++v) af.u[v] = half ? 0u : qr[v];
    }

    // ---- pass 1: row max of logits/temp -------------------------------
    float m8[8];
#pragma unroll
    for (int v = 0; v < 8; ++v) m8[v] = -1e30f;
    for (int j0 = 0; j0 < Tc; j0 += 16) {
        const v8f c = logits_tile(af, kbase, j0, n, half);
#pragma unroll
        for (int v = 0; v < 8; ++v) m8[v] = fmaxf(m8[v], c[v] * inv_temp);
    }
#pragma unroll
    for (int v = 0; v < 8; ++v)
        for (int off = 1; off < 16; off <<= 1)
            m8[v] = fmaxf(m8[v], __shfl_xor(m8[v], off, 32));

    // ---- pass 2: row sum of exp ---------------------------------------
    float l8[8];
#pragma unroll
    for (int v = 0; v < 8; ++v) l8[v] = 0.f;
    for (int j0 = 0; j0 < Tc; j0 += 16) {
        const v8f c = logits_tile(af, kbase, j0, n, half);
#pragma unroll
        for (int v = 0; v < 8; ++v) l8[v] += __expf(c[v] * inv_temp - m8[v]);
    }
#pragma unroll
    for (int v = 0; v < 8; ++v) {
        for (int off = 1; off < 16; off <<= 1)
            l8[v] += __shfl_xor(l8[v], off, 32);
        l8[v] = 1.f / l8[v];   // now 1/denominator
    }

    // ---- pass 3: stage attn tile + coalesced async drain + attn@V -----
    v8f o[4];
#pragma unroll
    for (int t = 0; t < 4; ++t) o[t] = (v8f){0.f,0.f,0.f,0.f,0.f,0.f,0.f,0.f};

    // cooperative drain mapping: thread -> (row, 16-float segment)
    const int drow = threadIdx.x >> 4;   // 0..15
    const int dseg = threadIdx.x & 15;   // 0..15
    // LDS byte offset of this thread's drain segment (flat addr low 32 bits
    // are the LDS offset per the aperture mapping rule)
    const uint32_t drain_lds =
        (uint32_t)(uintptr_t)&attn_stage[drow * 256 + dseg * 16];

    for (int j0 = 0; j0 < Tc; j0 += 32) {
        const v8f c0 = logits_tile(af, kbase, j0,      n, half);
        const v8f c1 = logits_tile(af, kbase, j0 + 16, n, half);
#pragma unroll
        for (int v = 0; v < 8; ++v) {
            const int r = v + 8 * half;                 // row within tile
            const float p0 = __expf(c0[v] * inv_temp - m8[v]) * l8[v];
            const float p1 = __expf(c1[v] * inv_temp - m8[v]) * l8[v];
            attn_stage[(r * 32 + n) * 8 + h]        = p0;
            attn_stage[(r * 32 + 16 + n) * 8 + h]   = p1;
            pshared[wave][r * 32 + n]      = (__bf16)p0;
            pshared[wave][r * 32 + 16 + n] = (__bf16)p1;
        }
        __syncthreads();

        // attn A-fragment from LDS (rows i0..i0+15, K = 32 j's)
        AB16 pa;
        const __bf16* ps = pshared[wave];
#pragma unroll
        for (int v = 0; v < 4; ++v) {
            pa.u[v]     = *(const uint32_t*)(ps + n * 32 + 16 * half + 2 * v);
            pa.u[v + 4] = *(const uint32_t*)(ps + n * 32 + 16 * half + 8 + 2 * v);
        }
        // V B-fragments: contiguous 32B loads from transposed bf16 V
#pragma unroll
        for (int ht = 0; ht < 4; ++ht) {
            AB16 vb;
            const uint32_t* vr = (const uint32_t*)(
                vbase + (size_t)(ht * 16 + n) * Tc + j0 + 16 * half);
#pragma unroll
            for (int v = 0; v < 8; ++v) vb.u[v] = vr[v];
            o[ht] = __builtin_amdgcn_wmma_f32_16x16x32_bf16(
                false, pa.v, false, vb.v, (short)0, o[ht], false, false);
        }

        // coalesced drain of the staged (16 x 32 x 8) fp32 attn tile:
        // 64B per thread via 4 async b128 LDS->global stores (offset is
        // applied to both the global and LDS address per the ISA).
        {
            const float* gaddr =
                attn + (((size_t)b * Tc + (i0 + drow)) * Tc + j0) * Hc + dseg * 16;
            asm volatile(
                "global_store_async_from_lds_b128 %0, %1, off\n\t"
                "global_store_async_from_lds_b128 %0, %1, off offset:16\n\t"
                "global_store_async_from_lds_b128 %0, %1, off offset:32\n\t"
                "global_store_async_from_lds_b128 %0, %1, off offset:48"
                :: "v"(gaddr), "v"(drain_lds) : "memory");
            asm volatile("s_wait_asynccnt 0x0" ::: "memory");
        }
        __syncthreads();   // staging tiles free to be overwritten
    }

    // store O (already normalized; attn was divided by denom before AV)
#pragma unroll
    for (int ht = 0; ht < 4; ++ht)
#pragma unroll
        for (int v = 0; v < 8; ++v) {
            const int row = i0 + v + 8 * half;
            const int col = h * HDc + ht * 16 + n;
            Oc[((size_t)b * Tc + row) * Dc + col] = o[ht][v];
        }
}

// ---------------------------------------------------------------------------
extern "C" void kernel_launch(void* const* d_in, const int* in_sizes, int n_in,
                              void* d_out, int out_size, void* d_ws, size_t ws_size,
                              hipStream_t stream) {
    const float* x             = (const float*)d_in[0];
    const float* qkv_w         = (const float*)d_in[1];
    const float* out_w         = (const float*)d_in[2];
    const float* base_centers  = (const float*)d_in[3];
    const float* center_deltas = (const float*)d_in[4];
    const float* log_scales    = (const float*)d_in[5];
    const float* log_amps      = (const float*)d_in[6];
    const float* movement      = (const float*)d_in[7];
    const float* temperature   = (const float*)d_in[8];

    float* out  = (float*)d_out;                       // (B,T,D)
    float* attn = out + (size_t)Bc * Tc * Dc;          // (B,T,T,H)

    // workspace carve
    char* w = (char*)d_ws;
    float*  qkv = (float*)w;  w += (size_t)Bc * Tc * 3 * Dc * sizeof(float);   // 25.2 MB
    __bf16* qwb = (__bf16*)w; w += (size_t)Bc * Hc * Tc * Sc * sizeof(__bf16); // 1 MB
    __bf16* kwb = (__bf16*)w; w += (size_t)Bc * Hc * Tc * Sc * sizeof(__bf16); // 1 MB
    __bf16* vtb = (__bf16*)w; w += (size_t)Bc * Hc * HDc * Tc * sizeof(__bf16);// 4 MB
    float*  Oc  = (float*)w;                                                   // 8 MB

    // 1) QKV projection: 4096x1536 <- 4096x512 @ (1536x512)^T
    {
        const int tiles = (Bc * Tc / 16) * (3 * Dc / 16);   // 24576
        gemm_bf16_kernel<<<tiles / 8, 256, 0, stream>>>(
            x, qkv_w, qkv, Bc * Tc, 3 * Dc, Dc);
    }
    // 2) Gaussian features + V transpose
    features_kernel<<<Bc * Tc, 128, 0, stream>>>(
        qkv, base_centers, center_deltas, log_scales, log_amps, movement,
        qwb, kwb, vtb);
    // 3) fused logits / softmax / attn-write / attn@V
    attn_kernel<<<Bc * (Tc / 16), 256, 0, stream>>>(
        qwb, kwb, vtb, temperature, attn, Oc);
    // 4) output projection: 4096x512 <- 4096x512 @ (512x512)^T
    {
        const int tiles = (Bc * Tc / 16) * (Dc / 16);       // 8192
        gemm_bf16_kernel<<<tiles / 8, 256, 0, stream>>>(
            Oc, out_w, out, Bc * Tc, Dc, Dc);
    }
}
